// SettleCharLM_13752485282435
// MI455X (gfx1250) — matrix operand order repaired and checked by reading, not compiled, not run on hardware
//
#include <hip/hip_runtime.h>

// ---------------- problem constants (from reference setup_inputs) -----------
#define DD    1024     // d
#define HH    4096     // 4*d
#define BB    32       // batch
#define TT    512      // timesteps
#define VV    256      // vocab
#define LL    4        // layers
#define NBLK  64       // persistent workgroups
#define NTHR  256      // threads per workgroup
#define NWAVE (NTHR / 32)
#define ALPHA_C 0.2f
#define EPS_C   1e-8f

typedef __attribute__((ext_vector_type(16))) __bf16 v16bf;
typedef __attribute__((ext_vector_type(8)))  __bf16 v8bf;
typedef __attribute__((ext_vector_type(8)))  float  v8f;

struct Ctx {
  // f32 inputs
  const int*   x;
  const float* tok_emb;
  const float* block_norm_scale;
  const float* fc1_b;
  const float* fc2_b;
  const float* out_norm_scale;
  const float* state_gate_b;
  const float* state_norm_scale;
  const float* settle_gate_b;
  // bf16 weights (converted into ws)
  const __bf16* fc1_w16;   // (L, HH, DD)
  const __bf16* fc2_w16;   // (L, DD, HH)
  const __bf16* sg_w16;    // (DD, DD)
  const __bf16* dg_w16;    // (DD, DD)
  const __bf16* head_w16;  // (VV, DD)
  // activations in ws
  float*  state;   // (BB, DD)
  float*  s_in;    // (BB, DD)  normalized state (f32, for gate epilogue)
  float*  h;       // (BB, DD)
  float*  prev;    // (BB, DD)
  __bf16* sA16;    // (BB, DD)  bf16 of s_in (gate GEMM A)
  __bf16* a16;     // (BB, DD)  bf16 normed activations (fc1 / head A)
  __bf16* hc16;    // (BB, DD)  bf16 core output (settle-gate A)
  __bf16* hh16;    // (BB, HH)  bf16 gelu(fc1) (fc2 A)
  float*  dAcc;    // (2,)
  unsigned* bar;   // grid barrier counter
  float*  out;     // d_out: logits (BB,TT,VV) then deltas (2,)
};

// ---------------- device helpers -------------------------------------------
__device__ __forceinline__ float sigmf(float x) { return 1.0f / (1.0f + __expf(-x)); }

__device__ __forceinline__ void grid_sync(unsigned* bar, unsigned epoch) {
  __syncthreads();
  if (threadIdx.x == 0) {
    __builtin_amdgcn_fence(__ATOMIC_RELEASE, "agent");
    atomicAdd(bar, 1u);
    const unsigned target = epoch * (unsigned)NBLK;
    while (atomicAdd(bar, 0u) < target) __builtin_amdgcn_s_sleep(1);
  }
  __syncthreads();
  __builtin_amdgcn_fence(__ATOMIC_ACQUIRE, "agent");
}

// One block per row (blocks >= BB fall through). rmsnorm(x)*scale -> bf16 (+f32).
__device__ __forceinline__ void rmsnorm_rows(const float* __restrict__ src,
                                             const float* __restrict__ scale,
                                             float* __restrict__ dst32,
                                             __bf16* __restrict__ dst16,
                                             float* red) {
  const int row = blockIdx.x;
  if (row >= BB) return;
  const float4 v = reinterpret_cast<const float4*>(src + row * DD)[threadIdx.x];
  red[threadIdx.x] = v.x * v.x + v.y * v.y + v.z * v.z + v.w * v.w;
  __syncthreads();
  for (int o = NTHR / 2; o > 0; o >>= 1) {
    if ((int)threadIdx.x < o) red[threadIdx.x] += red[threadIdx.x + o];
    __syncthreads();
  }
  const float r = rsqrtf(red[0] * (1.0f / (float)DD) + EPS_C);
  __syncthreads();  // red[] reused by next phase
  const float4 sc = reinterpret_cast<const float4*>(scale)[threadIdx.x];
  float4 o4;
  o4.x = v.x * r * sc.x; o4.y = v.y * r * sc.y;
  o4.z = v.z * r * sc.z; o4.w = v.w * r * sc.w;
  if (dst32) reinterpret_cast<float4*>(dst32 + row * DD)[threadIdx.x] = o4;
  __bf16* d = dst16 + row * DD + threadIdx.x * 4;
  d[0] = (__bf16)o4.x; d[1] = (__bf16)o4.y; d[2] = (__bf16)o4.z; d[3] = (__bf16)o4.w;
}

// EPI: 0=state gate, 1=fc1+gelu, 2=fc2 residual, 3=settle gate, 4=head
// C = A(32xK bf16) x W(NxK bf16)^T, one 16x16 tile per SPLIT-wave group.
// SPLIT>1: waves (wid%SPLIT) each do a K/SPLIT slice; LDS-reduce; slice-0 wave
// runs the epilogue. Tile->block mapping keeps __syncthreads convergent.
template <int EPI, int SPLIT>
__device__ __forceinline__ void run_gemm(const Ctx& c,
                                         const __bf16* __restrict__ A, const int K,
                                         const __bf16* __restrict__ W,
                                         const float* __restrict__ bias,
                                         const int ntiles, const int t, const int kidx,
                                         const int wid, const int lane,
                                         float* __restrict__ part) {
  const int lt    = (SPLIT == 1) ? wid : (wid / SPLIT);
  const int slice = (SPLIT == 1) ? 0 : (wid % SPLIT);
  const int tpb   = NWAVE / SPLIT;        // tiles per block per trip
  const int klen  = K / SPLIT;
  for (int tile = blockIdx.x * tpb + lt; tile < ntiles; tile += NBLK * tpb) {
    const int m_base = (tile & 1) << 4;
    const int n_base = (tile >> 1) << 4;
    // A frag: lane -> M = lane&15 ; 16 contiguous K at (lane&16)?16:0
    const int am = m_base + (lane & 15);
    const int ak = ((lane & 16) ? 16 : 0) + slice * klen;
    // B frag: lane -> N = lane&15 ; 16 contiguous K at (lane&16)?16:0
    const int bn = n_base + (lane & 15);
    const int bk = ((lane & 16) ? 16 : 0) + slice * klen;
    const __bf16* __restrict__ ap = A + (size_t)am * K + ak;
    const __bf16* __restrict__ bp = W + (size_t)bn * K + bk;
    v8f acc = {};
    for (int kb = 0; kb < klen; kb += 32) {
      v8bf lo = *reinterpret_cast<const v8bf*>(ap + kb);
      v8bf hi = *reinterpret_cast<const v8bf*>(ap + kb + 8);
      v16bf a;
#pragma unroll
      for (int i = 0; i < 8; ++i) { a[i] = lo[i]; a[i + 8] = hi[i]; }
      v16bf b = *reinterpret_cast<const v16bf*>(bp + kb);
      acc = __builtin_amdgcn_wmma_f32_16x16x32_bf16(false, a, false, b,
                                                    (short)0, acc, false, false);
    }
    if (SPLIT > 1) {
      float* my = part + (size_t)(wid * 32 + lane) * 8;
      if (slice != 0) {
#pragma unroll
        for (int i = 0; i < 8; ++i) my[i] = acc[i];
      }
      __syncthreads();
      if (slice == 0) {
#pragma unroll
        for (int s = 1; s < SPLIT; ++s) {
          const float* p = part + (size_t)((wid + s) * 32 + lane) * 8;
#pragma unroll
          for (int i = 0; i < 8; ++i) acc[i] += p[i];
        }
      }
      __syncthreads();
      if (slice != 0) continue;   // epilogue is slice-0 only
    }
    // epilogue: element r -> (m = m_base + r + ((lane&16)?8:0), n = n_base + (lane&15))
    const int n     = n_base + (lane & 15);
    const int mrow0 = (lane & 16) ? 8 : 0;
    const float bval = bias ? bias[n] : 0.0f;
    float dsum = 0.0f;
#pragma unroll
    for (int r = 0; r < 8; ++r) {
      const int m = m_base + mrow0 + r;
      float v = acc[r] + bval;
      if (EPI == 0) {            // g = sigmoid(.); h = emb_t + g*s_in ; prev = h
        const float g = sigmf(v);
        const int tok = c.x[m * TT + t];
        const float hv = c.tok_emb[(size_t)tok * DD + n] + g * c.s_in[m * DD + n];
        c.h[m * DD + n] = hv;
        c.prev[m * DD + n] = hv;
      } else if (EPI == 1) {     // exact GELU -> hh16
        const float gl = 0.5f * v * (1.0f + erff(v * 0.70710678f));
        c.hh16[m * HH + n] = (__bf16)gl;
      } else if (EPI == 2) {     // residual add -> h (f32) and hc16 (bf16)
        const float hv = c.h[m * DD + n] + v;
        c.h[m * DD + n] = hv;
        c.hc16[m * DD + n] = (__bf16)hv;
      } else if (EPI == 3) {     // h = prev + dg*(hc-prev); prev = h; |delta| acc
        const float dg  = sigmf(v);
        const float hc  = c.h[m * DD + n];
        const float pv  = c.prev[m * DD + n];
        const float dlt = dg * (hc - pv);
        const float hn2 = pv + dlt;
        c.h[m * DD + n] = hn2;
        c.prev[m * DD + n] = hn2;
        dsum += fabsf(dlt);
      } else {                   // head: logits[b=m, t, v=n]
        c.out[(size_t)m * ((size_t)TT * VV) + (size_t)t * VV + n] = v;
      }
    }
    if (EPI == 3) {
#pragma unroll
      for (int off = 16; off > 0; off >>= 1) dsum += __shfl_xor(dsum, off, 32);
      if (lane == 0) atomicAdd(&c.dAcc[kidx], dsum * (1.0f / ((float)BB * (float)DD)));
    }
  }
}

// ---------------- kernels ---------------------------------------------------
__global__ void cvt_f32_bf16(const float* __restrict__ src, __bf16* __restrict__ dst, int n) {
  int i = blockIdx.x * blockDim.x + threadIdx.x;
  const int stride = gridDim.x * blockDim.x;
  for (; i < n; i += stride) dst[i] = (__bf16)src[i];
}

__global__ void init_kernel(unsigned* bar, float* dAcc, float* state, float* s_in,
                            __bf16* sA16) {
  const int i = blockIdx.x * blockDim.x + threadIdx.x;
  if (i == 0) *bar = 0u;
  if (i < 2) dAcc[i] = 0.0f;
  for (int j = i; j < BB * DD; j += gridDim.x * blockDim.x) {
    state[j] = 0.0f;
    s_in[j] = 0.0f;
    sA16[j] = (__bf16)0.0f;
  }
}

__global__ __launch_bounds__(NTHR, 1) void settle_persistent(Ctx c) {
  __shared__ float red[NTHR];
  __shared__ float part[NWAVE * 32 * 8];     // split-K partial accumulators (8 KB)
  const int lane = threadIdx.x & 31;
  const int wid  = threadIdx.x >> 5;
  unsigned epoch = 0;
#define GSYNC() do { ++epoch; grid_sync(c.bar, epoch); } while (0)

  for (int t = 0; t < TT; ++t) {
    // P1: gate GEMM (split-K4) + h = emb_t + g*s_in ; prev = h
    run_gemm<0, 4>(c, c.sA16, DD, c.sg_w16, c.state_gate_b, 2 * (DD / 16), t, 0,
                   wid, lane, part);
    GSYNC();

    for (int ks = 0; ks < 2; ++ks) {              // K_SETTLE
      for (int l = 0; l < LL; ++l) {              // core: residual MLP blocks
        rmsnorm_rows(c.h, c.block_norm_scale + l * DD, nullptr, c.a16, red);
        GSYNC();
        run_gemm<1, 1>(c, c.a16, DD, c.fc1_w16 + (size_t)l * HH * DD, c.fc1_b + l * HH,
                       2 * (HH / 16), t, 0, wid, lane, part);
        GSYNC();
        run_gemm<2, 4>(c, c.hh16, HH, c.fc2_w16 + (size_t)l * DD * HH, c.fc2_b + l * DD,
                       2 * (DD / 16), t, 0, wid, lane, part);
        GSYNC();
      }
      // settle gate (split-K4): h = prev + dg*(hc - prev); prev = h; deltas
      run_gemm<3, 4>(c, c.hc16, DD, c.dg_w16, c.settle_gate_b, 2 * (DD / 16), t, ks,
                     wid, lane, part);
      GSYNC();
    }

    // PF: blocks 0..31 -> out rmsnorm into a16 (head A)
    //     blocks 32..63 -> state EMA + state rmsnorm -> s_in (f32) + sA16 (bf16)
    if (blockIdx.x < BB) {
      rmsnorm_rows(c.h, c.out_norm_scale, nullptr, c.a16, red);
    } else {
      const int row = blockIdx.x - BB;            // one state row per block
      float4 st = reinterpret_cast<float4*>(c.state + row * DD)[threadIdx.x];
      const float4 hv = reinterpret_cast<const float4*>(c.h + row * DD)[threadIdx.x];
      st.x = (1.0f - ALPHA_C) * st.x + ALPHA_C * hv.x;
      st.y = (1.0f - ALPHA_C) * st.y + ALPHA_C * hv.y;
      st.z = (1.0f - ALPHA_C) * st.z + ALPHA_C * hv.z;
      st.w = (1.0f - ALPHA_C) * st.w + ALPHA_C * hv.w;
      reinterpret_cast<float4*>(c.state + row * DD)[threadIdx.x] = st;
      red[threadIdx.x] = st.x * st.x + st.y * st.y + st.z * st.z + st.w * st.w;
      __syncthreads();
      for (int o = NTHR / 2; o > 0; o >>= 1) {
        if ((int)threadIdx.x < o) red[threadIdx.x] += red[threadIdx.x + o];
        __syncthreads();
      }
      const float r = rsqrtf(red[0] * (1.0f / (float)DD) + EPS_C);
      __syncthreads();
      const float4 sc = reinterpret_cast<const float4*>(c.state_norm_scale)[threadIdx.x];
      float4 si;
      si.x = st.x * r * sc.x; si.y = st.y * r * sc.y;
      si.z = st.z * r * sc.z; si.w = st.w * r * sc.w;
      reinterpret_cast<float4*>(c.s_in + row * DD)[threadIdx.x] = si;
      __bf16* d = c.sA16 + row * DD + threadIdx.x * 4;
      d[0] = (__bf16)si.x; d[1] = (__bf16)si.y; d[2] = (__bf16)si.z; d[3] = (__bf16)si.w;
    }
    GSYNC();
    // head (split-K4, blocks 0..15): logits_t = a16 @ head_w^T
    run_gemm<4, 4>(c, c.a16, DD, c.head_w16, nullptr, 2 * (VV / 16), t, 0,
                   wid, lane, part);
    GSYNC();
  }
  // finalize delta_per_k = sum_t mean|.| / T
  if (blockIdx.x == 0 && threadIdx.x < 2)
    c.out[(size_t)BB * TT * VV + threadIdx.x] = c.dAcc[threadIdx.x] * (1.0f / (float)TT);
#undef GSYNC
}

// ---------------- host ------------------------------------------------------
extern "C" void kernel_launch(void* const* d_in, const int* in_sizes, int n_in,
                              void* d_out, int out_size, void* d_ws, size_t ws_size,
                              hipStream_t stream) {
  const int*   x       = (const int*)d_in[0];
  const float* tok_emb = (const float*)d_in[1];
  const float* bns     = (const float*)d_in[2];
  const float* fc1_w   = (const float*)d_in[3];
  const float* fc1_b   = (const float*)d_in[4];
  const float* fc2_w   = (const float*)d_in[5];
  const float* fc2_b   = (const float*)d_in[6];
  const float* ons     = (const float*)d_in[7];
  const float* head_w  = (const float*)d_in[8];
  const float* sg_w    = (const float*)d_in[9];
  const float* sg_b    = (const float*)d_in[10];
  const float* sns     = (const float*)d_in[11];
  const float* dg_w    = (const float*)d_in[12];
  const float* dg_b    = (const float*)d_in[13];

  char* w = (char*)d_ws;
  size_t o = 0;
  auto alloc = [&](size_t bytes) -> void* {
    o = (o + 255) & ~(size_t)255;
    void* p = (void*)(w + o);
    o += bytes;
    return p;
  };
  unsigned* bar   = (unsigned*)alloc(256);
  float*    dAcc  = (float*)alloc(256);
  float*    state = (float*)alloc((size_t)BB * DD * 4);
  float*    s_in  = (float*)alloc((size_t)BB * DD * 4);
  float*    h     = (float*)alloc((size_t)BB * DD * 4);
  float*    prev  = (float*)alloc((size_t)BB * DD * 4);
  __bf16*   sA16  = (__bf16*)alloc((size_t)BB * DD * 2);
  __bf16*   a16   = (__bf16*)alloc((size_t)BB * DD * 2);
  __bf16*   hc16  = (__bf16*)alloc((size_t)BB * DD * 2);
  __bf16*   hh16  = (__bf16*)alloc((size_t)BB * HH * 2);
  __bf16*   f1w   = (__bf16*)alloc((size_t)LL * HH * DD * 2);
  __bf16*   f2w   = (__bf16*)alloc((size_t)LL * DD * HH * 2);
  __bf16*   sgw   = (__bf16*)alloc((size_t)DD * DD * 2);
  __bf16*   dgw   = (__bf16*)alloc((size_t)DD * DD * 2);
  __bf16*   hw    = (__bf16*)alloc((size_t)VV * DD * 2);

  init_kernel<<<64, 256, 0, stream>>>(bar, dAcc, state, s_in, sA16);
  cvt_f32_bf16<<<2048, 256, 0, stream>>>(fc1_w, f1w, LL * HH * DD);
  cvt_f32_bf16<<<2048, 256, 0, stream>>>(fc2_w, f2w, LL * DD * HH);
  cvt_f32_bf16<<<512, 256, 0, stream>>>(sg_w, sgw, DD * DD);
  cvt_f32_bf16<<<512, 256, 0, stream>>>(dg_w, dgw, DD * DD);
  cvt_f32_bf16<<<128, 256, 0, stream>>>(head_w, hw, VV * DD);

  Ctx c;
  c.x = x; c.tok_emb = tok_emb; c.block_norm_scale = bns;
  c.fc1_b = fc1_b; c.fc2_b = fc2_b; c.out_norm_scale = ons;
  c.state_gate_b = sg_b; c.state_norm_scale = sns; c.settle_gate_b = dg_b;
  c.fc1_w16 = f1w; c.fc2_w16 = f2w; c.sg_w16 = sgw; c.dg_w16 = dgw; c.head_w16 = hw;
  c.state = state; c.s_in = s_in; c.h = h; c.prev = prev;
  c.sA16 = sA16; c.a16 = a16; c.hc16 = hc16; c.hh16 = hh16;
  c.dAcc = dAcc; c.bar = bar; c.out = (float*)d_out;

  settle_persistent<<<NBLK, NTHR, 0, stream>>>(c);

  (void)in_sizes; (void)n_in; (void)out_size; (void)ws_size;
}
